// ALIGNN_46540265619596
// MI455X (gfx1250) — matrix operand order repaired and checked by reading, not compiled, not run on hardware
//
#include <hip/hip_runtime.h>
#include <cstdint>
#include <cstddef>

// ---------------------------------------------------------------------------
// ALIGNN forward for MI455X (gfx1250, wave32).
// - All 256x256 projections on v_wmma_f32_16x16x32_bf16 (bf16 A/B, f32 acc).
// - E2-row edge-gate GEMM fused with gather/sigmoid/atomic-scatter/LayerNorm.
// - A-tiles staged with GLOBAL_LOAD_ASYNC_TO_LDS_B128 (ASYNCcnt); phase-B
//   gather rows prefetched behind the GEMM.
// ---------------------------------------------------------------------------

#define Hd   256
#define Nn   2048
#define Ne   24576
#define Ne2  294912

typedef __attribute__((ext_vector_type(16))) __bf16       bf16x16;
typedef __attribute__((ext_vector_type(8)))  float        f32x8;
typedef __attribute__((ext_vector_type(4)))  unsigned int u32x4;
typedef int v4i_gcc __attribute__((vector_size(16)));     // builtin's vector flavor

union AFrag { bf16x16 v; u32x4 q[2]; };

#if __has_builtin(__builtin_amdgcn_global_load_async_to_lds_b128) && \
    __has_builtin(__builtin_amdgcn_s_wait_asynccnt)
#define HAVE_ASYNC_LDS 1
#else
#define HAVE_ASYNC_LDS 0
#endif

#if HAVE_ASYNC_LDS
__device__ __forceinline__ void async_b128(const void* g, void* l) {
  void* gq = (void*)(uintptr_t)g;  // strip const; flat==global numerically
  __builtin_amdgcn_global_load_async_to_lds_b128(
      (__attribute__((address_space(1))) v4i_gcc*)gq,
      (__attribute__((address_space(3))) v4i_gcc*)l, 0, 0);
}
#endif

__device__ __forceinline__ unsigned short f2bf(float f) {
  unsigned int u = __float_as_uint(f);
  u += 0x7FFFu + ((u >> 16) & 1u);          // round-to-nearest-even
  return (unsigned short)(u >> 16);
}

__device__ __forceinline__ float wred32(float v) {
#pragma unroll
  for (int m = 16; m > 0; m >>= 1) v += __shfl_xor(v, m, 32);
  return v;
}

__device__ __forceinline__ float sigm(float v) { return 1.f / (1.f + __expf(-v)); }
__device__ __forceinline__ float silu(float v) { return v * sigm(v); }

// --------------------------- weight prep (f32 [K][N] -> bf16 [N][K]) -------
__global__ __launch_bounds__(256) void wprep(const float* __restrict__ W,
                                             unsigned short* __restrict__ Wt) {
  int t = blockIdx.x * 256 + threadIdx.x;   // 65536 elements
  int k = t >> 8, n = t & 255;
  Wt[(size_t)n * Hd + k] = f2bf(W[t]);
}

// --------------------------- embedding lookup ------------------------------
__global__ __launch_bounds__(256) void embed(const int* __restrict__ an,
                                             const float* __restrict__ emb,
                                             float* __restrict__ x) {
  int i = blockIdx.x, c = threadIdx.x;
  x[(size_t)i * Hd + c] = emb[(size_t)an[i] * Hd + c];
}

// --------------------------- WMMA 32x256 @ 256x256 core --------------------
// wave tile: 16 rows x 64 cols (4 accumulators). 8 waves = 2 row x 4 col blocks.
__device__ __forceinline__ void wmma_tile_256(const unsigned short* ldsA,
                                              const unsigned short* __restrict__ Wt,
                                              int rb, int cb, int lane,
                                              f32x8 acc[4]) {
  const int lhalf = lane >> 4, l16 = lane & 15;
#pragma unroll
  for (int kk = 0; kk < 8; ++kk) {
    const int kb = kk * 32;
    AFrag a;
    const unsigned short* arow = ldsA + (rb * 16 + l16) * Hd;
    a.q[0] = *(const u32x4*)(arow + kb + lhalf * 8);        // K = kb+{0..7|8..15}
    a.q[1] = *(const u32x4*)(arow + kb + 16 + lhalf * 8);   // K = kb+16+{...}
#pragma unroll
    for (int j = 0; j < 4; ++j) {
      const int col = cb * 64 + j * 16 + l16;
      AFrag b;
      const unsigned short* wrow = Wt + (size_t)col * Hd + kb + lhalf * 8;
      b.q[0] = *(const u32x4*)(wrow);
      b.q[1] = *(const u32x4*)(wrow + 16);
      acc[j] = __builtin_amdgcn_wmma_f32_16x16x32_bf16(
          false, a.v, false, b.v, (short)0, acc[j], false, false);
    }
  }
}

// --------------------------- generic GEMM + bias ---------------------------
// dynamic LDS: [0,16K) bf16 A-tile; [16K,48K) f32 async staging buffer.
__global__ __launch_bounds__(256) void gemm256_bias(
    const float* __restrict__ A, const unsigned short* __restrict__ Wt,
    const float* __restrict__ bias, float* __restrict__ out) {
  extern __shared__ char smem[];
  unsigned short* ldsA = (unsigned short*)smem;             // 16 KB
  float* ldsF = (float*)(smem + 32 * Hd * 2);               // 32 KB
  const int row0 = blockIdx.x * 32;
  const int tid = threadIdx.x;
#if HAVE_ASYNC_LDS
  // 256 lanes x 16B (one b128 each, 128B apart), straight into LDS (ASYNCcnt)
  async_b128(A + (size_t)row0 * Hd + tid * 32, ldsF + tid * 32);
  __builtin_amdgcn_s_wait_asynccnt(0);
  __syncthreads();
  for (int i = 0; i < 32; ++i)
    ldsA[i * Hd + tid] = f2bf(ldsF[i * Hd + tid]);
#else
  for (int i = 0; i < 32; ++i)
    ldsA[i * Hd + tid] = f2bf(A[(size_t)(row0 + i) * Hd + tid]);
#endif
  __syncthreads();
  const int wave = tid >> 5, lane = tid & 31;
  const int rb = wave >> 2, cb = wave & 3;
  const f32x8 zero = {0.f, 0.f, 0.f, 0.f, 0.f, 0.f, 0.f, 0.f};
  f32x8 acc[4];
  for (int j = 0; j < 4; ++j) acc[j] = zero;
  wmma_tile_256(ldsA, Wt, rb, cb, lane, acc);
  const int lhalf = lane >> 4, l16 = lane & 15;
  for (int j = 0; j < 4; ++j) {
    const int col = cb * 64 + j * 16 + l16;
    const float bv = bias[col];
#pragma unroll
    for (int i = 0; i < 8; ++i)
      out[(size_t)(row0 + rb * 16 + lhalf * 8 + i) * Hd + col] = acc[j][i] + bv;
  }
}

// --------------------------- fused EGGC edge pass --------------------------
// e = Y@Weg + beg + XS[src] + XD[dst]; sigma = cut*sigmoid(e);
// atomic scatter num/den by dst; Y += silu(LN(e)).  In-place on Y.
// dynamic LDS: [0,16K) bf16 A-tile; [16K,48K) f32 staging, reused for e-tile.
__global__ __launch_bounds__(256) void eggc_edge(
    float* __restrict__ Y, const unsigned short* __restrict__ WegT,
    const float* __restrict__ beg,
    const float* __restrict__ XS, const float* __restrict__ XD,
    const float* __restrict__ BH,
    const int* __restrict__ src, const int* __restrict__ dst,
    const float* __restrict__ cutoff,
    float* __restrict__ num, float* __restrict__ den,
    const float* __restrict__ lng, const float* __restrict__ lnb) {
  extern __shared__ char smem[];
  unsigned short* ldsA = (unsigned short*)smem;             // 16 KB
  float* ldsE = (float*)(smem + 32 * Hd * 2);               // 32 KB
  const int row0 = blockIdx.x * 32;
  const int tid = threadIdx.x;
  const int wave = tid >> 5, lane = tid & 31;
  // prefetch the random-access gather rows this wave scatters in phase B;
  // their latency hides behind the WMMA GEMM.
  for (int rr = wave * 4; rr < wave * 4 + 4; ++rr) {
    const int gr = row0 + rr;
    const size_t sb = (size_t)src[gr] * Hd, db = (size_t)dst[gr] * Hd;
    __builtin_prefetch(XS + sb + lane * 8, 0, 3);
    __builtin_prefetch(XD + db + lane * 8, 0, 3);
    __builtin_prefetch(BH + sb + lane * 8, 0, 3);
  }
#if HAVE_ASYNC_LDS
  async_b128(Y + (size_t)row0 * Hd + tid * 32, ldsE + tid * 32);
  __builtin_amdgcn_s_wait_asynccnt(0);
  __syncthreads();
  for (int i = 0; i < 32; ++i)
    ldsA[i * Hd + tid] = f2bf(ldsE[i * Hd + tid]);
#else
  for (int i = 0; i < 32; ++i)
    ldsA[i * Hd + tid] = f2bf(Y[(size_t)(row0 + i) * Hd + tid]);
#endif
  __syncthreads();
  const int rb = wave >> 2, cb = wave & 3;
  const f32x8 zero = {0.f, 0.f, 0.f, 0.f, 0.f, 0.f, 0.f, 0.f};
  f32x8 acc[4];
  for (int j = 0; j < 4; ++j) acc[j] = zero;
  wmma_tile_256(ldsA, WegT, rb, cb, lane, acc);
  const int lhalf = lane >> 4, l16 = lane & 15;
  for (int j = 0; j < 4; ++j) {
    const int col = cb * 64 + j * 16 + l16;
    const float bv = beg[col];
#pragma unroll
    for (int i = 0; i < 8; ++i)
      ldsE[(rb * 16 + lhalf * 8 + i) * Hd + col] = acc[j][i] + bv;
  }
  __syncthreads();
  // phase B: 8 waves x 4 rows each
  for (int rr = wave * 4; rr < wave * 4 + 4; ++rr) {
    const int gr = row0 + rr;
    const int s = src[gr], dd = dst[gr];
    const float cut = cutoff ? cutoff[gr] : 1.0f;
    const size_t sb = (size_t)s * Hd, db = (size_t)dd * Hd;
    float ev[8];
    float sum = 0.f, sq = 0.f;
#pragma unroll
    for (int t = 0; t < 8; ++t) {
      const int c = lane * 8 + t;
      const float e = ldsE[rr * Hd + c] + XS[sb + c] + XD[db + c];
      ev[t] = e;
      const float sg = cut * sigm(e);
      atomicAdd(&den[db + c], sg);
      atomicAdd(&num[db + c], sg * BH[sb + c]);
      sum += e; sq += e * e;
    }
    sum = wred32(sum); sq = wred32(sq);
    const float mu = sum * (1.f / 256.f);
    const float var = sq * (1.f / 256.f) - mu * mu;
    const float rstd = rsqrtf(var + 1e-5f);
#pragma unroll
    for (int t = 0; t < 8; ++t) {
      const int c = lane * 8 + t;
      const float v = (ev[t] - mu) * rstd * lng[c] + lnb[c];
      Y[(size_t)gr * Hd + c] += silu(v);
    }
  }
}

// --------------------------- node update: X += silu(LN(XSU + num/den)) -----
__global__ __launch_bounds__(256) void node_update(
    float* __restrict__ X, const float* __restrict__ XSU,
    const float* __restrict__ num, const float* __restrict__ den,
    const float* __restrict__ lng, const float* __restrict__ lnb) {
  const int wave = threadIdx.x >> 5, lane = threadIdx.x & 31;
  const size_t base = (size_t)(blockIdx.x * 8 + wave) * Hd;
  float v[8];
  float sum = 0.f, sq = 0.f;
#pragma unroll
  for (int t = 0; t < 8; ++t) {
    const int c = lane * 8 + t;
    const float h = num[base + c] / (den[base + c] + 1e-6f);
    const float u = XSU[base + c] + h;
    v[t] = u; sum += u; sq += u * u;
  }
  sum = wred32(sum); sq = wred32(sq);
  const float mu = sum * (1.f / 256.f);
  const float var = sq * (1.f / 256.f) - mu * mu;
  const float rstd = rsqrtf(var + 1e-5f);
#pragma unroll
  for (int t = 0; t < 8; ++t) {
    const int c = lane * 8 + t;
    const float u = (v[t] - mu) * rstd * lng[c] + lnb[c];
    X[base + c] += silu(u);
  }
}

// --------------------------- edge features (RBF80 -> 64 -> 256) ------------
__global__ __launch_bounds__(256) void edge_feats(
    const float* __restrict__ r,
    const float* __restrict__ w1, const float* __restrict__ b1,
    const float* __restrict__ gA, const float* __restrict__ bA,
    const float* __restrict__ w2, const float* __restrict__ b2,
    const float* __restrict__ gB, const float* __restrict__ bB,
    float* __restrict__ y, float* __restrict__ cutoff) {
  __shared__ float rbf[8][80];
  __shared__ float h1[8][64];
  const int wave = threadIdx.x >> 5, lane = threadIdx.x & 31;
  const int ei = blockIdx.x * 8 + wave;
  const float rx = r[ei * 3], ry = r[ei * 3 + 1], rz = r[ei * 3 + 2];
  const float dd = sqrtf(rx * rx + ry * ry + rz * rz);
  const float step = 8.0f / 79.0f, gamma = 79.0f / 8.0f;
  for (int i = lane; i < 80; i += 32) {
    const float t = dd - step * i;
    rbf[wave][i] = __expf(-gamma * t * t);
  }
  __syncthreads();
  float o1[2];
#pragma unroll
  for (int j = 0; j < 2; ++j) {
    const int o = lane * 2 + j;
    float s = b1[o];
    for (int i = 0; i < 80; ++i) s += rbf[wave][i] * w1[i * 64 + o];
    o1[j] = s;
  }
  float sum = wred32(o1[0] + o1[1]);
  float sq  = wred32(o1[0] * o1[0] + o1[1] * o1[1]);
  float mu = sum * (1.f / 64.f), var = sq * (1.f / 64.f) - mu * mu;
  float rstd = rsqrtf(var + 1e-5f);
#pragma unroll
  for (int j = 0; j < 2; ++j) {
    const int o = lane * 2 + j;
    h1[wave][o] = silu((o1[j] - mu) * rstd * gA[o] + bA[o]);
  }
  __syncthreads();
  float o2[8];
#pragma unroll
  for (int j = 0; j < 8; ++j) {
    const int o = lane * 8 + j;
    float s = b2[o];
    for (int i = 0; i < 64; ++i) s += h1[wave][i] * w2[i * 256 + o];
    o2[j] = s;
  }
  sum = 0.f; sq = 0.f;
#pragma unroll
  for (int j = 0; j < 8; ++j) { sum += o2[j]; sq += o2[j] * o2[j]; }
  sum = wred32(sum); sq = wred32(sq);
  mu = sum * (1.f / 256.f); var = sq * (1.f / 256.f) - mu * mu;
  rstd = rsqrtf(var + 1e-5f);
#pragma unroll
  for (int j = 0; j < 8; ++j) {
    const int o = lane * 8 + j;
    y[(size_t)ei * Hd + o] = silu((o2[j] - mu) * rstd * gB[o] + bB[o]);
  }
  if (lane == 0) {
    const float r2 = dd * dd, ri2 = 56.25f, ro2 = 64.f;
    const float s = (ro2 - r2) * (ro2 - r2) * (ro2 + 2.f * r2 - 3.f * ri2)
                    * (1.f / (7.75f * 7.75f * 7.75f));
    cutoff[ei] = dd < 7.5f ? 1.f : (dd < 8.f ? s : 0.f);
  }
}

// --------------------------- angle features (RBF40 -> 64 -> 256) -----------
__global__ __launch_bounds__(256) void angle_feats(
    const float* __restrict__ r,
    const int* __restrict__ lgs, const int* __restrict__ lgd,
    const float* __restrict__ w1, const float* __restrict__ b1,
    const float* __restrict__ gA, const float* __restrict__ bA,
    const float* __restrict__ w2, const float* __restrict__ b2,
    const float* __restrict__ gB, const float* __restrict__ bB,
    float* __restrict__ z) {
  __shared__ float rbf[8][40];
  __shared__ float h1[8][64];
  const int wave = threadIdx.x >> 5, lane = threadIdx.x & 31;
  const int k = blockIdx.x * 8 + wave;
  const int ls = lgs[k], ld = lgd[k];
  const float ax = r[ls * 3], ay = r[ls * 3 + 1], az = r[ls * 3 + 2];
  const float bx = r[ld * 3], by = r[ld * 3 + 1], bz = r[ld * 3 + 2];
  const float dot = -(ax * bx + ay * by + az * bz);
  const float na = sqrtf(ax * ax + ay * ay + az * az);
  const float nb = sqrtf(bx * bx + by * by + bz * bz);
  const float cs = fminf(1.f, fmaxf(-1.f, dot / (na * nb)));
  const float step = 2.0f / 39.0f, gamma = 39.0f / 2.0f;
  for (int i = lane; i < 40; i += 32) {
    const float t = cs - (-1.0f + step * i);
    rbf[wave][i] = __expf(-gamma * t * t);
  }
  __syncthreads();
  float o1[2];
#pragma unroll
  for (int j = 0; j < 2; ++j) {
    const int o = lane * 2 + j;
    float s = b1[o];
    for (int i = 0; i < 40; ++i) s += rbf[wave][i] * w1[i * 64 + o];
    o1[j] = s;
  }
  float sum = wred32(o1[0] + o1[1]);
  float sq  = wred32(o1[0] * o1[0] + o1[1] * o1[1]);
  float mu = sum * (1.f / 64.f), var = sq * (1.f / 64.f) - mu * mu;
  float rstd = rsqrtf(var + 1e-5f);
#pragma unroll
  for (int j = 0; j < 2; ++j) {
    const int o = lane * 2 + j;
    h1[wave][o] = silu((o1[j] - mu) * rstd * gA[o] + bA[o]);
  }
  __syncthreads();
  float o2[8];
#pragma unroll
  for (int j = 0; j < 8; ++j) {
    const int o = lane * 8 + j;
    float s = b2[o];
    for (int i = 0; i < 64; ++i) s += h1[wave][i] * w2[i * 256 + o];
    o2[j] = s;
  }
  sum = 0.f; sq = 0.f;
#pragma unroll
  for (int j = 0; j < 8; ++j) { sum += o2[j]; sq += o2[j] * o2[j]; }
  sum = wred32(sum); sq = wred32(sq);
  mu = sum * (1.f / 256.f); var = sq * (1.f / 256.f) - mu * mu;
  rstd = rsqrtf(var + 1e-5f);
#pragma unroll
  for (int j = 0; j < 8; ++j) {
    const int o = lane * 8 + j;
    z[(size_t)k * Hd + o] = silu((o2[j] - mu) * rstd * gB[o] + bB[o]);
  }
}

// --------------------------- mean pool + fc --------------------------------
__global__ __launch_bounds__(256) void final_pool(
    const float* __restrict__ X, const float* __restrict__ fcw,
    const float* __restrict__ fcb, float* __restrict__ out) {
  __shared__ float red[256];
  const int c = threadIdx.x;
  float s = 0.f;
  for (int i = 0; i < Nn; ++i) s += X[(size_t)i * Hd + c];
  red[c] = (s * (1.f / (float)Nn)) * fcw[c];
  __syncthreads();
  for (int off = 128; off > 0; off >>= 1) {
    if (c < off) red[c] += red[c + off];
    __syncthreads();
  }
  if (c == 0) out[0] = red[0] + fcb[0];
}

// ===========================================================================
// Host launch
// ===========================================================================
extern "C" void kernel_launch(void* const* d_in, const int* in_sizes, int n_in,
                              void* d_out, int out_size, void* d_ws, size_t ws_size,
                              hipStream_t stream) {
  (void)in_sizes; (void)n_in; (void)out_size; (void)ws_size;
  auto P = [&](int i) { return (const float*)d_in[i]; };
  const int* an  = (const int*)d_in[0];
  const int* src = (const int*)d_in[1];
  const int* dst = (const int*)d_in[2];
  const float* r = (const float*)d_in[3];
  const int* lgs = (const int*)d_in[4];
  const int* lgd = (const int*)d_in[5];

  // pytree (sorted-key) flattening of params:
  //   alignn[i]: edge eggc @ 6+i*28, node eggc @ 6+i*28+14 (14 leaves each)
  //   eggc offsets: 0 dg.b 1 dg.w 2 du.b 3 du.w 4 eg.b 5 eg.w
  //                 6 lne.b 7 lne.g 8 lnn.b 9 lnn.g 10 sg.b 11 sg.w 12 su.b 13 su.w
  const int ANG1_B = 118, ANG1_W = 119, ANG2_B = 120, ANG2_W = 121;
  const int ANGLN1_B = 122, ANGLN1_G = 123, ANGLN2_B = 124, ANGLN2_G = 125;
  const int ATOM_EMB = 126;
  const int E1_B = 127, E1_W = 128, E2_B = 129, E2_W = 130;
  const int ELN1_B = 131, ELN1_G = 132, ELN2_B = 133, ELN2_G = 134;
  const int FC_B = 135, FC_W = 136, GCN_BASE = 137;

  // workspace carve (256B aligned)
  char* ws = (char*)d_ws;
  auto carve = [&](size_t bytes) {
    char* p = ws; ws += (bytes + 255) & ~(size_t)255; return p;
  };
  float* x   = (float*)carve((size_t)Nn * Hd * 4);
  float* y   = (float*)carve((size_t)Ne * Hd * 4);
  float* z   = (float*)carve((size_t)Ne2 * Hd * 4);
  float* XS  = (float*)carve((size_t)Ne * Hd * 4);
  float* XD  = (float*)carve((size_t)Ne * Hd * 4);
  float* BH  = (float*)carve((size_t)Ne * Hd * 4);
  float* XSU = (float*)carve((size_t)Ne * Hd * 4);
  float* num = (float*)carve((size_t)Ne * Hd * 4);
  float* den = (float*)carve((size_t)Ne * Hd * 4);
  float* cut = (float*)carve((size_t)Ne * 4);
  unsigned short* wtPool = (unsigned short*)carve((size_t)60 * Hd * Hd * 2);

  // 12 EGGCs in execution order: (node, edge) x4 alignn, then 4 gcn
  int eggcBase[12];
  for (int i = 0; i < 4; ++i) {
    eggcBase[2 * i]     = 6 + i * 28 + 14;  // node
    eggcBase[2 * i + 1] = 6 + i * 28;       // edge
  }
  for (int i = 0; i < 4; ++i) eggcBase[8 + i] = GCN_BASE + i * 14;

  // transpose + bf16-convert all 60 projection weights once (L2-resident)
  const int wslot[5] = {11, 1, 3, 13, 5};   // sg.w dg.w du.w su.w eg.w
  for (int o = 0; o < 12; ++o)
    for (int s = 0; s < 5; ++s)
      wprep<<<256, 256, 0, stream>>>(P(eggcBase[o] + wslot[s]),
                                     wtPool + ((size_t)o * 5 + s) * Hd * Hd);

  embed<<<Nn, 256, 0, stream>>>(an, P(ATOM_EMB), x);
  edge_feats<<<Ne / 8, 256, 0, stream>>>(r, P(E1_W), P(E1_B), P(ELN1_G), P(ELN1_B),
                                         P(E2_W), P(E2_B), P(ELN2_G), P(ELN2_B),
                                         y, cut);
  angle_feats<<<Ne2 / 8, 256, 0, stream>>>(r, lgs, lgd,
                                           P(ANG1_W), P(ANG1_B), P(ANGLN1_G), P(ANGLN1_B),
                                           P(ANG2_W), P(ANG2_B), P(ANGLN2_G), P(ANGLN2_B),
                                           z);

  // NOTE: 48KB dynamic LDS is reserved unconditionally (16KB A-tile + 32KB
  // staging/e-tile) so host/device disagreement on HAVE_ASYNC_LDS is safe.
  const size_t kLds = (size_t)32 * Hd * 2 + (size_t)32 * Hd * 4;

  auto run_eggc = [&](int pb, int ord, float* Xn, float* Yed,
                      const int* s, const int* d, const float* cutp,
                      int n_nodes, int n_edges) {
    const unsigned short* wt5 = wtPool + (size_t)ord * 5 * Hd * Hd;
    gemm256_bias<<<n_nodes / 32, 256, kLds, stream>>>(Xn, wt5 + 0 * Hd * Hd, P(pb + 10), XS);
    gemm256_bias<<<n_nodes / 32, 256, kLds, stream>>>(Xn, wt5 + 1 * Hd * Hd, P(pb + 0),  XD);
    gemm256_bias<<<n_nodes / 32, 256, kLds, stream>>>(Xn, wt5 + 2 * Hd * Hd, P(pb + 2),  BH);
    gemm256_bias<<<n_nodes / 32, 256, kLds, stream>>>(Xn, wt5 + 3 * Hd * Hd, P(pb + 12), XSU);
    (void)hipMemsetAsync(num, 0, (size_t)n_nodes * Hd * 4, stream);
    (void)hipMemsetAsync(den, 0, (size_t)n_nodes * Hd * 4, stream);
    eggc_edge<<<n_edges / 32, 256, kLds, stream>>>(
        Yed, wt5 + 4 * Hd * Hd, P(pb + 4), XS, XD, BH, s, d, cutp,
        num, den, P(pb + 7), P(pb + 6));
    node_update<<<n_nodes / 8, 256, 0, stream>>>(Xn, XSU, num, den,
                                                 P(pb + 9), P(pb + 8));
  };

  for (int i = 0; i < 4; ++i) {
    run_eggc(eggcBase[2 * i],     2 * i,     x, y, src, dst, cut, Nn, Ne);      // node conv
    run_eggc(eggcBase[2 * i + 1], 2 * i + 1, y, z, lgs, lgd, nullptr, Ne, Ne2); // edge conv
  }
  for (int i = 0; i < 4; ++i)
    run_eggc(eggcBase[8 + i], 8 + i, x, y, src, dst, cut, Nn, Ne);

  final_pool<<<1, 256, 0, stream>>>(x, P(FC_W), P(FC_B), (float*)d_out);
}
